// TransformerDecodeLevel_88407606820964
// MI455X (gfx1250) — matrix operand order repaired and checked by reading, not compiled, not run on hardware
//
#include <hip/hip_runtime.h>
#include <hip/hip_bf16.h>

typedef _Float16 half_t;
typedef __attribute__((ext_vector_type(16))) _Float16 v16h;
typedef __attribute__((ext_vector_type(8)))  _Float16 v8h;
typedef __attribute__((ext_vector_type(8)))  float    v8f;

// ---------------------------------------------------------------------------
// helpers
// ---------------------------------------------------------------------------
__device__ __forceinline__ v8f zero8() {
  v8f r;
#pragma unroll
  for (int i = 0; i < 8; i++) r[i] = 0.f;
  return r;
}

__device__ __forceinline__ v16h zero16() {
  v16h r;
#pragma unroll
  for (int i = 0; i < 16; i++) r[i] = (half_t)0.f;
  return r;
}

// B-tile fragment: lane-half g holds K = g*16 + {0..15} (contiguous 16 halfs).
// Caller passes per-lane pointer already offset by (lane&15)*stride + (lane>>4)*16.
__device__ __forceinline__ v16h ldB(const half_t* p) {
  v8h a = *(const v8h*)p;
  v8h b = *(const v8h*)(p + 8);
  v16h r;
#pragma unroll
  for (int i = 0; i < 8; i++) { r[i] = a[i]; r[i + 8] = b[i]; }
  return r;
}

// A-tile fragment: lane-half g holds K = g*16 + {0..15} (contiguous 16 halfs).
// Caller passes row base pointer (lane&15)*stride (+ k0), plus hl = lane>>4.
__device__ __forceinline__ v16h ldA(const half_t* p, int hl) {
  v8h a = *(const v8h*)(p + hl * 16);
  v8h b = *(const v8h*)(p + hl * 16 + 8);
  v16h r;
#pragma unroll
  for (int i = 0; i < 8; i++) { r[i] = a[i]; r[i + 8] = b[i]; }
  return r;
}

__device__ __forceinline__ v8f wmma_f16(v16h a, v16h b, v8f c) {
#if defined(__HIP_DEVICE_COMPILE__)
  return __builtin_amdgcn_wmma_f32_16x16x32_f16(false, a, false, b, (short)0, c,
                                                false, false);
#else
  return c;
#endif
}

// Async global -> LDS 16-byte copy (ASYNCcnt tracked). Generic LDS pointers on
// gfx1250 carry the LDS byte offset in their low 32 bits.
__device__ __forceinline__ void async_ld16(half_t* lds, const half_t* g) {
#if defined(__HIP_DEVICE_COMPILE__)
  unsigned l = (unsigned)(unsigned long long)(uintptr_t)lds;
  unsigned long long ga = (unsigned long long)(uintptr_t)g;
  asm volatile("global_load_async_to_lds_b128 %0, %1, off"
               :: "v"(l), "v"(ga) : "memory");
#endif
}

__device__ __forceinline__ void wait_async0() {
#if defined(__HIP_DEVICE_COMPILE__)
  asm volatile("s_wait_asynccnt 0x0" ::: "memory");
#endif
}

// ---------------------------------------------------------------------------
// weight / input conversion kernels
// ---------------------------------------------------------------------------
__global__ void k_cvt(const float* __restrict__ in, half_t* __restrict__ out, int n) {
  int i = blockIdx.x * 256 + threadIdx.x;
  if (i < n) out[i] = (half_t)in[i];
}

// ConvTranspose weight (Cin=512, Co=128, K, K) -> [tap][co][ci] f16
__global__ void k_cvt_convw(const float* __restrict__ w, half_t* __restrict__ out,
                            int K, int n) {
  int idx = blockIdx.x * 256 + threadIdx.x;
  if (idx >= n) return;
  int ci = idx & 511;
  int co = (idx >> 9) & 127;
  int tap = idx >> 16;
  int kh = tap / K, kw = tap % K;
  out[idx] = (half_t)w[(((size_t)ci * 128 + co) * K + kh) * K + kw];
}

// x (2,512,64,128) NCHW f32 -> xh (2*64*128, 512) token-major f16
__global__ void k_tr_x(const float* __restrict__ x, half_t* __restrict__ xh) {
  int i = blockIdx.x * 256 + threadIdx.x;
  if (i >= 2 * 512 * 64 * 128) return;
  int c = i & 511;
  int t = i >> 9;
  int b = t >> 13;
  int hw = t & 8191;
  xh[i] = (half_t)x[((size_t)(b * 512 + c)) * 8192 + hw];
}

// ---------------------------------------------------------------------------
// ConvTranspose (both paths) -> d_out (B,256,128,256) NCHW f32
// one wave per 16(out-ch) x 16(pixel) tile inside a parity class
// ---------------------------------------------------------------------------
__global__ void __launch_bounds__(256) k_convt(
    const half_t* __restrict__ xh, const half_t* __restrict__ w2t,
    const half_t* __restrict__ w4t, const float* __restrict__ b2,
    const float* __restrict__ b4, float* __restrict__ out) {
  const int lane = threadIdx.x & 31;
  const int n = lane & 15, hl = lane >> 4;
  int tid = blockIdx.x * 8 + (threadIdx.x >> 5);
  int ct = tid & 15;
  int pw = (tid >> 4) & 1;
  int wg = (tid >> 5) & 7;
  int ho = (tid >> 8) & 127;
  int b = tid >> 15;
  v8f acc = zero8();
  int wo = wg * 32 + pw + 2 * n;
  if (ct < 8) {  // 2x2 path: output channels 0..127
    int co = ct * 16;
    int tap = (ho & 1) * 2 + pw;
    int hi = ho >> 1;
    const half_t* Ap = w2t + ((size_t)tap * 128 + co + n) * 512;
    const half_t* Bp =
        xh + ((size_t)((b * 64 + hi) * 128 + wg * 16 + n)) * 512 + hl * 16;
    __builtin_prefetch(Ap, 0, 0);
#pragma unroll 4
    for (int k0 = 0; k0 < 512; k0 += 32)
      acc = wmma_f16(ldA(Ap + k0, hl), ldB(Bp + k0), acc);
#pragma unroll
    for (int r = 0; r < 8; r++) {
      int cof = co + r + 8 * hl;
      out[(((size_t)b * 256 + cof) * 128 + ho) * 256 + wo] = acc[r] + b2[cof];
    }
  } else {  // 4x4 path: output channels 128..255, 4 masked taps
    int cw = (ct - 8) * 16;
    int p = ho & 1, q = pw;
    int hb = (ho + p) >> 1;
    int wb = wg * 16 + pw + n;
    for (int a = 0; a < 2; a++) {
      int hi = hb - a;
      int kh = (1 - p) + 2 * a;
      bool okh = (hi >= 0) && (hi < 64);
      for (int c2 = 0; c2 < 2; c2++) {
        int kw = (1 - q) + 2 * c2;
        int wi = wb - c2;
        bool ok = okh && (wi >= 0) && (wi < 128);
        const half_t* Ap = w4t + ((size_t)(kh * 4 + kw) * 128 + cw + n) * 512;
        const half_t* Bp =
            xh + ((size_t)((b * 64 + hi) * 128 + wi)) * 512 + hl * 16;
#pragma unroll 4
        for (int k0 = 0; k0 < 512; k0 += 32) {
          v16h bt = ok ? ldB(Bp + k0) : zero16();
          acc = wmma_f16(ldA(Ap + k0, hl), bt, acc);
        }
      }
    }
#pragma unroll
    for (int r = 0; r < 8; r++) {
      int cof = 128 + cw + r + 8 * hl;
      out[(((size_t)b * 256 + cof) * 128 + ho) * 256 + wo] =
          acc[r] + b4[cw + r + 8 * hl];
    }
  }
}

// ---------------------------------------------------------------------------
// channel LayerNorm: h (NCHW f32) -> xn (token-major f16), optional roll by -shift
// one wave per pixel
// ---------------------------------------------------------------------------
__global__ void __launch_bounds__(256) k_ln(const float* __restrict__ h,
                                            const float* __restrict__ g,
                                            const float* __restrict__ bb,
                                            half_t* __restrict__ xn, int shift) {
  int lane = threadIdx.x & 31;
  int pix = blockIdx.x * 8 + (threadIdx.x >> 5);
  int b = pix >> 15;
  int rem = pix & 32767;
  int y = rem >> 8, x = rem & 255;
  const float* base = h + (size_t)b * 8388608 + y * 256 + x;
  float v[8];
  float s = 0.f, s2 = 0.f;
#pragma unroll
  for (int j = 0; j < 8; j++) {
    int c = lane + 32 * j;
    float t = base[(size_t)c * 32768];
    v[j] = t;
    s += t;
    s2 += t * t;
  }
#pragma unroll
  for (int m = 16; m >= 1; m >>= 1) {
    s += __shfl_xor(s, m);
    s2 += __shfl_xor(s2, m);
  }
  float mu = s * (1.f / 256.f);
  float var = s2 * (1.f / 256.f) - mu * mu;
  float rs = rsqrtf(var + 1e-5f);
  int yd = (y - shift) & 127, xd = (x - shift) & 255;
  half_t* dst = xn + ((size_t)b * 32768 + yd * 256 + xd) * 256;
#pragma unroll
  for (int j = 0; j < 8; j++) {
    int c = lane + 32 * j;
    dst[c] = (half_t)((v[j] - mu) * rs * g[c] + bb[c]);
  }
}

// ---------------------------------------------------------------------------
// windowed attention: one workgroup per 8x8 window, one head per wave
// ---------------------------------------------------------------------------
#define XWS 264   // staged xn tile row stride (256 + pad)
#define QKS 520   // q,k row stride (512 + pad)
#define VTS 72    // vT row stride (64 tokens + pad)
#define SBS 72    // softmax row stride
#define OBS 264   // O row stride (256 + pad)

__global__ void __launch_bounds__(256) k_attn(
    const half_t* __restrict__ xn, const half_t* __restrict__ wqkv,
    const half_t* __restrict__ wout, const float* __restrict__ bout,
    float* __restrict__ out, int shift) {
  extern __shared__ char smem[];
  half_t* xw = (half_t*)smem;        // [64][264] : staged window activations
  half_t* qk = xw + 64 * XWS;        // [64][520] : q cols 0..255, k cols 256..511
  half_t* vT = qk + 64 * QKS;        // [256][72] : V transposed
  half_t* sb0 = vT + 256 * VTS;      // 8 waves x [64][72]
  half_t* ob = sb0 + 8 * 64 * SBS;   // [64][264]

  const int lane = threadIdx.x & 31;
  const int wv = threadIdx.x >> 5;
  const int n = lane & 15, hl = lane >> 4;
  int w = blockIdx.x;
  int b = w >> 9;
  int r9 = w & 511;
  int y0 = (r9 >> 5) * 8;
  int x0 = (r9 & 31) * 8;

  // ---- phase 0: async-stage the 64x256 window tile into LDS ----
#pragma unroll
  for (int c = 0; c < 8; ++c) {
    int id = threadIdx.x + 256 * c;  // 2048 chunks of 16B
    int wt = id >> 5;                // token in window
    int ch = id & 31;                // 16B chunk within 512B row
    int gy = y0 + (wt >> 3), gx = x0 + (wt & 7);
    const half_t* gsrc = xn + ((size_t)b * 32768 + gy * 256 + gx) * 256 + ch * 8;
    async_ld16(xw + (size_t)wt * XWS + ch * 8, gsrc);
  }
  wait_async0();
  __syncthreads();

  // ---- phase 1: qkv = xn_window @ wqkv^T  (192 tiles, 24 per wave) ----
  for (int it = 0; it < 24; ++it) {
    int t = wv * 24 + it;
    int ocb = (t >> 2) * 16;
    int tokb = (t & 3) * 16;
    const half_t* Bp = xw + (size_t)(tokb + n) * XWS + hl * 16;
    const half_t* Ap = wqkv + (size_t)(ocb + n) * 256;
    __builtin_prefetch(Ap, 0, 0);
    v8f acc = zero8();
#pragma unroll
    for (int k0 = 0; k0 < 256; k0 += 32)
      acc = wmma_f16(ldA(Ap + k0, hl), ldB(Bp + k0), acc);
    if (ocb < 512) {
#pragma unroll
      for (int r = 0; r < 8; r++)
        qk[(size_t)(tokb + n) * QKS + ocb + r + 8 * hl] = (half_t)acc[r];
    } else {
#pragma unroll
      for (int r = 0; r < 8; r++)
        vT[(size_t)(ocb - 512 + r + 8 * hl) * VTS + tokb + n] = (half_t)acc[r];
    }
  }
  __syncthreads();

  // ---- phase 2: RoPE on q and k (in LDS) ----
  for (int t = 0; t < 64; t++) {
    int idx = threadIdx.x + 256 * t;
    int dp = idx & 15;
    int hh = (idx >> 4) & 7;
    int wt = (idx >> 7) & 63;
    int reg = (idx >> 13) & 1;
    int pos = (dp < 8) ? (wt >> 3) : (wt & 7);
    int f = dp & 7;
    float th = (float)pos * __expf(-(float)f * 1.1512925465f);  // ln(1e4)/8
    float cs = __cosf(th), sn = __sinf(th);
    half_t* pr = qk + (size_t)wt * QKS + reg * 256 + hh * 32 + dp;
    float x1 = (float)pr[0], x2 = (float)pr[16];
    pr[0] = (half_t)(x1 * cs - x2 * sn);
    pr[16] = (half_t)(x2 * cs + x1 * sn);
  }
  __syncthreads();

  // ---- phase 3: S = softmax(scale * Q K^T), one head per wave ----
  half_t* sb = sb0 + (size_t)wv * 64 * SBS;
  const int h32 = wv * 32;
  const float scale = 0.17677669529663687f;  // 32^-0.5
  for (int qi = 0; qi < 4; qi++) {
    v8f st[4];
    const half_t* Aq = qk + (size_t)(qi * 16 + n) * QKS + h32;
#pragma unroll
    for (int kj = 0; kj < 4; kj++) {
      const half_t* Bk = qk + (size_t)(kj * 16 + n) * QKS + 256 + h32 + hl * 16;
      st[kj] = wmma_f16(ldA(Aq, hl), ldB(Bk), zero8());
    }
#pragma unroll
    for (int r = 0; r < 8; r++) {
      float a0 = st[0][r] * scale, a1 = st[1][r] * scale;
      float a2 = st[2][r] * scale, a3 = st[3][r] * scale;
      float mx = fmaxf(fmaxf(a0, a1), fmaxf(a2, a3));
      for (int m = 8; m >= 1; m >>= 1) mx = fmaxf(mx, __shfl_xor(mx, m));
      float e0 = __expf(a0 - mx), e1 = __expf(a1 - mx);
      float e2 = __expf(a2 - mx), e3 = __expf(a3 - mx);
      float sm = e0 + e1 + e2 + e3;
      for (int m = 8; m >= 1; m >>= 1) sm += __shfl_xor(sm, m);
      float inv = 1.f / sm;
      int row = qi * 16 + r + 8 * hl;
      sb[(size_t)row * SBS + 0 + n] = (half_t)(e0 * inv);
      sb[(size_t)row * SBS + 16 + n] = (half_t)(e1 * inv);
      sb[(size_t)row * SBS + 32 + n] = (half_t)(e2 * inv);
      sb[(size_t)row * SBS + 48 + n] = (half_t)(e3 * inv);
    }
  }

  // ---- phase 4: O = P V  -> ob [token][256] ----
  for (int qi = 0; qi < 4; qi++) {
    const half_t* Ap = sb + (size_t)(qi * 16 + n) * SBS;
    for (int dt = 0; dt < 2; dt++) {
      const half_t* Bp = vT + (size_t)(h32 + dt * 16 + n) * VTS + hl * 16;
      v8f acc = zero8();
      acc = wmma_f16(ldA(Ap, hl), ldB(Bp), acc);
      acc = wmma_f16(ldA(Ap + 32, hl), ldB(Bp + 32), acc);
#pragma unroll
      for (int r = 0; r < 8; r++)
        ob[(size_t)(qi * 16 + r + 8 * hl) * OBS + h32 + dt * 16 + n] =
            (half_t)acc[r];
    }
  }
  __syncthreads();

  // ---- phase 5: out-proj + residual add (64 tiles, 8 per wave) ----
  for (int it = 0; it < 8; ++it) {
    int t = wv * 8 + it;
    int cot = t >> 2;
    int tokt = t & 3;
    const half_t* Ap = wout + (size_t)(cot * 16 + n) * 256;
    const half_t* Bp = ob + (size_t)(tokt * 16 + n) * OBS + hl * 16;
    __builtin_prefetch(Ap, 0, 0);
    v8f acc = zero8();
#pragma unroll
    for (int k0 = 0; k0 < 256; k0 += 32)
      acc = wmma_f16(ldA(Ap + k0, hl), ldB(Bp + k0), acc);
    int wt = tokt * 16 + n;
    int gy = (y0 + (wt >> 3) + shift) & 127;
    int gx = (x0 + (wt & 7) + shift) & 255;
#pragma unroll
    for (int r = 0; r < 8; r++) {
      int co = cot * 16 + r + 8 * hl;
      size_t o = (((size_t)b * 256 + co) * 128 + gy) * 256 + gx;
      out[o] += acc[r] + bout[co];
    }
  }
}

// ---------------------------------------------------------------------------
// fused feed-forward: hidden tile kept in LDS, one workgroup per 64 tokens
// ---------------------------------------------------------------------------
#define HIDS 1032
__global__ void __launch_bounds__(256) k_ff(
    const half_t* __restrict__ xn, const half_t* __restrict__ w1,
    const float* __restrict__ b1, const half_t* __restrict__ w2,
    const float* __restrict__ b2, float* __restrict__ out) {
  extern __shared__ char smem[];
  half_t* xw = (half_t*)smem;    // [64][264] staged activations
  half_t* hid = xw + 64 * XWS;   // [64][1032]
  const int lane = threadIdx.x & 31;
  const int wv = threadIdx.x >> 5;
  const int n = lane & 15, hl = lane >> 4;
  int t0 = blockIdx.x * 64;
  int b = t0 >> 15;
  int rem = t0 & 32767;
  int y = rem >> 8, xq = rem & 255;

  // async-stage the 64x256 token tile (contiguous in xn) into LDS
#pragma unroll
  for (int c = 0; c < 8; ++c) {
    int id = threadIdx.x + 256 * c;  // 2048 chunks of 16B
    int wt = id >> 5;
    int ch = id & 31;
    async_ld16(xw + (size_t)wt * XWS + ch * 8,
               xn + (size_t)(t0 + wt) * 256 + ch * 8);
  }
  wait_async0();
  __syncthreads();

  // GEMM1 + bias + gelu -> LDS  (256 tiles, 32 per wave)
  for (int it = 0; it < 32; ++it) {
    int t = wv * 32 + it;
    int oct = t >> 2, tokt = t & 3;
    const half_t* Ap = w1 + (size_t)(oct * 16 + n) * 256;
    const half_t* Bp = xw + (size_t)(tokt * 16 + n) * XWS + hl * 16;
    __builtin_prefetch(Ap, 0, 0);
    v8f acc = zero8();
#pragma unroll
    for (int k0 = 0; k0 < 256; k0 += 32)
      acc = wmma_f16(ldA(Ap + k0, hl), ldB(Bp + k0), acc);
    int token = tokt * 16 + n;
    int oc0 = oct * 16 + 8 * hl;
    v8h pk;
#pragma unroll
    for (int r = 0; r < 8; r++) {
      float xg = acc[r] + b1[oc0 + r];
      float inner = 0.7978845608028654f * (xg + 0.044715f * xg * xg * xg);
      pk[r] = (half_t)(0.5f * xg * (1.f + tanhf(inner)));
    }
    *(v8h*)(hid + (size_t)token * HIDS + oc0) = pk;
  }
  __syncthreads();

  // GEMM2 + bias + residual add  (64 tiles, 8 per wave, K=1024)
  for (int it = 0; it < 8; ++it) {
    int t = wv * 8 + it;
    int cot = t >> 2, tokt = t & 3;
    const half_t* Ap = w2 + (size_t)(cot * 16 + n) * 1024;
    const half_t* Bp = hid + (size_t)(tokt * 16 + n) * HIDS + hl * 16;
    __builtin_prefetch(Ap, 0, 0);
    v8f acc = zero8();
#pragma unroll 4
    for (int k0 = 0; k0 < 1024; k0 += 32)
      acc = wmma_f16(ldA(Ap + k0, hl), ldB(Bp + k0), acc);
    int x = xq + tokt * 16 + n;
#pragma unroll
    for (int r = 0; r < 8; r++) {
      int co = cot * 16 + r + 8 * hl;
      size_t o = (((size_t)b * 256 + co) * 128 + y) * 256 + x;
      out[o] += acc[r] + b2[co];
    }
  }
}

// ---------------------------------------------------------------------------
// host orchestration
// ---------------------------------------------------------------------------
extern "C" void kernel_launch(void* const* d_in, const int* in_sizes, int n_in,
                              void* d_out, int out_size, void* d_ws,
                              size_t ws_size, hipStream_t stream) {
  const float* x = (const float*)d_in[0];
  const float* ct_w2 = (const float*)d_in[1];
  const float* ct_b2 = (const float*)d_in[2];
  const float* ct_w4 = (const float*)d_in[3];
  const float* ct_b4 = (const float*)d_in[4];
  const float* a1_g = (const float*)d_in[5];
  const float* a1_b = (const float*)d_in[6];
  const float* a1_wqkv = (const float*)d_in[7];
  const float* a1_wout = (const float*)d_in[8];
  const float* a1_bout = (const float*)d_in[9];
  const float* a2_g = (const float*)d_in[10];
  const float* a2_b = (const float*)d_in[11];
  const float* a2_wqkv = (const float*)d_in[12];
  const float* a2_wout = (const float*)d_in[13];
  const float* a2_bout = (const float*)d_in[14];
  const float* f1_g = (const float*)d_in[15];
  const float* f1_b = (const float*)d_in[16];
  const float* f1_w1 = (const float*)d_in[17];
  const float* f1_b1 = (const float*)d_in[18];
  const float* f1_w2 = (const float*)d_in[19];
  const float* f1_b2 = (const float*)d_in[20];
  const float* f2_g = (const float*)d_in[21];
  const float* f2_b = (const float*)d_in[22];
  const float* f2_w1 = (const float*)d_in[23];
  const float* f2_b1 = (const float*)d_in[24];
  const float* f2_w2 = (const float*)d_in[25];
  const float* f2_b2 = (const float*)d_in[26];
  float* out = (float*)d_out;

  half_t* xh = (half_t*)d_ws;          // 2*64*128 x 512
  half_t* xn = xh + 8388608;           // 65536 x 256
  half_t* w2t = xn + 16777216;         // 4 x 128 x 512
  half_t* w4t = w2t + 262144;          // 16 x 128 x 512
  half_t* qkv1 = w4t + 1048576;        // 768 x 256
  half_t* wo1 = qkv1 + 196608;         // 256 x 256
  half_t* w1a = wo1 + 65536;           // 1024 x 256
  half_t* w2a = w1a + 262144;          // 256 x 1024
  half_t* qkv2 = w2a + 262144;
  half_t* wo2 = qkv2 + 196608;
  half_t* w1b = wo2 + 65536;
  half_t* w2b = w1b + 262144;

  const int ATTN_LDS =
      (64 * XWS + 64 * QKS + 256 * VTS + 8 * 64 * SBS + 64 * OBS) * 2;
  const int FF_LDS = (64 * XWS + 64 * HIDS) * 2;
  hipFuncSetAttribute(reinterpret_cast<const void*>(k_attn),
                      hipFuncAttributeMaxDynamicSharedMemorySize, ATTN_LDS);
  hipFuncSetAttribute(reinterpret_cast<const void*>(k_ff),
                      hipFuncAttributeMaxDynamicSharedMemorySize, FF_LDS);

  // weight conversion to f16
  k_cvt_convw<<<262144 / 256, 256, 0, stream>>>(ct_w2, w2t, 2, 262144);
  k_cvt_convw<<<1048576 / 256, 256, 0, stream>>>(ct_w4, w4t, 4, 1048576);
  k_cvt<<<196608 / 256, 256, 0, stream>>>(a1_wqkv, qkv1, 196608);
  k_cvt<<<65536 / 256, 256, 0, stream>>>(a1_wout, wo1, 65536);
  k_cvt<<<262144 / 256, 256, 0, stream>>>(f1_w1, w1a, 262144);
  k_cvt<<<262144 / 256, 256, 0, stream>>>(f1_w2, w2a, 262144);
  k_cvt<<<196608 / 256, 256, 0, stream>>>(a2_wqkv, qkv2, 196608);
  k_cvt<<<65536 / 256, 256, 0, stream>>>(a2_wout, wo2, 65536);
  k_cvt<<<262144 / 256, 256, 0, stream>>>(f2_w1, w1b, 262144);
  k_cvt<<<262144 / 256, 256, 0, stream>>>(f2_w2, w2b, 262144);

  // input layout transform, conv-transpose, then the 4 residual blocks
  k_tr_x<<<8388608 / 256, 256, 0, stream>>>(x, xh);
  k_convt<<<8192, 256, 0, stream>>>(xh, w2t, w4t, ct_b2, ct_b4, out);

  k_ln<<<8192, 256, 0, stream>>>(out, a1_g, a1_b, xn, 0);
  k_attn<<<1024, 256, ATTN_LDS, stream>>>(xn, qkv1, wo1, a1_bout, out, 0);

  k_ln<<<8192, 256, 0, stream>>>(out, f1_g, f1_b, xn, 0);
  k_ff<<<1024, 256, FF_LDS, stream>>>(xn, w1a, f1_b1, w2a, f1_b2, out);

  k_ln<<<8192, 256, 0, stream>>>(out, a2_g, a2_b, xn, 4);
  k_attn<<<1024, 256, ATTN_LDS, stream>>>(xn, qkv2, wo2, a2_bout, out, 4);

  k_ln<<<8192, 256, 0, stream>>>(out, f2_g, f2_b, xn, 0);
  k_ff<<<1024, 256, FF_LDS, stream>>>(xn, w1b, f2_b1, w2b, f2_b2, out);
}
